// SSD_81570018886356
// MI455X (gfx1250) — matrix operand order repaired and checked by reading, not compiled, not run on hardware
//
#include <hip/hip_runtime.h>
#include <hip/hip_bf16.h>

typedef __attribute__((ext_vector_type(16))) _Float16 v16h;
typedef __attribute__((ext_vector_type(8)))  _Float16 v8h;
typedef __attribute__((ext_vector_type(8)))  float    v8f;

#define NB 4
#define NUMPRIORS 8732
#define NCLS 21
#define TOPK 200
#define CONF_TH 0.01f
#define NMS_TH 0.45f

// ---------------------------------------------------------------------------
// fp32 -> fp16 weight packing with padded row stride Ks (multiple of 32)
// ---------------------------------------------------------------------------
__global__ void pack_w16_kernel(const float* __restrict__ src,
                                _Float16* __restrict__ dst,
                                int Cout, int K, int Ks) {
  int i = blockIdx.x * 256 + threadIdx.x;
  int n = Cout * Ks;
  if (i >= n) return;
  int co = i / Ks, k = i - co * Ks;
  dst[i] = (k < K) ? (_Float16)src[(long long)co * K + k] : (_Float16)0.f;
}

// ---------------------------------------------------------------------------
// Implicit-GEMM convolution with WMMA (fp16 in, fp32 accumulate).
// Block = 128 threads (4 wave32) computing a 64(Cout) x 64(pixel) macro-tile.
// Waves 2x2; each wave register-blocks 2 A-frags x 2 B-frags -> 4 WMMA/chunk.
// LDS tiles are stored "fragment-ready":
//   A: half-index ((row*2 + hi)*16 + m), kk = 16*hi + m
//      -> a lane's 16 halves are one aligned 32B vector.
//   B: column-major Bs[col*32 + kk] -> one aligned 32B vector per fragment.
// Weight rows padded to Ks (mult of 32) so A staging is a single aligned
// 32B global vector load per thread, no bounds checks.
// Optional NHWC-concat epilogue for the SSD heads.
// ---------------------------------------------------------------------------
__global__ __launch_bounds__(128)
void conv_wmma_kernel(const float* __restrict__ in,
                      const _Float16* __restrict__ w16,
                      const float* __restrict__ bias,
                      float* __restrict__ out,
                      int B, int Cin, int H, int W,
                      int Cout, int Ho, int Wo,
                      int kh, int kw, int stride, int pad, int dil, int relu,
                      int Kw /* padded weight row stride */,
                      int nhwcTotal, int nhwcOff) {
  __shared__ _Float16 As[64 * 32];   // fragment-ready A (64 rows x 32 k)
  __shared__ _Float16 Bs[64 * 32];   // column-major B   (64 cols x 32 k)

  const int tid  = threadIdx.x;
  const int lane = tid & 31;
  const int wave = tid >> 5;
  const int waveM = (wave >> 1) * 32;   // wave's cout offset within macro-tile
  const int waveN = (wave & 1) * 32;    // wave's pixel offset within macro-tile
  const int coutBase = blockIdx.x * 64;
  const int pixBase  = blockIdx.y * 64;
  const int npix = B * Ho * Wo;
  const int HoWo = Ho * Wo;
  const int khw  = kh * kw;
  const int K    = Cin * khw;

  // per-thread staging assignments (fixed across the K loop)
  const int arow = tid >> 1;            // A row (cout within tile), 0..63
  const int hsel = (tid & 1) << 3;      // A m-offset: 0 or 8 (from kk half)
  const int kkb  = (tid & 1) << 4;      // k sub-chunk: 0 or 16
  const int bcol = tid >> 1;            // B col (pixel within tile), 0..63

  // hoisted pixel decode for B staging (once, not per chunk)
  const int bpix = pixBase + bcol;
  int boy = 0, box = 0;
  long long binbase = 0;
  const bool bvalid = bpix < npix;
  if (bvalid) {
    int b   = bpix / HoWo;
    int rem = bpix - b * HoWo;
    boy = rem / Wo;
    box = rem - boy * Wo;
    binbase = (long long)b * Cin * H * W;
  }
  const int aco = coutBase + arow;
  const bool avalid = aco < Cout;

  v8f accs[2][2] = {};

  for (int k0 = 0; k0 < K; k0 += 32) {
    // prefetch next A chunk (global_prefetch_b8)
    if (avalid && k0 + 32 < K)
      __builtin_prefetch(w16 + (long long)aco * Kw + k0 + 32, 0, 3);

    // --- stage A: one aligned 32B global load + two 16B LDS stores ---
    {
      v16h wv = {};
      if (avalid)
        wv = *(const v16h*)(w16 + (long long)aco * Kw + k0 + kkb);
      v8h lo = __builtin_shufflevector(wv, wv, 0, 1, 2, 3, 4, 5, 6, 7);
      v8h hi = __builtin_shufflevector(wv, wv, 8, 9, 10, 11, 12, 13, 14, 15);
      *(v8h*)(As + (arow << 5) + kkb)      = lo;  // (row), k = kkb..kkb+7
      *(v8h*)(As + (arow << 5) + kkb + 8)  = hi;  // (row), k = kkb+8..kkb+15
    }
    // --- stage B: 16-k gather strip per thread, one aligned 32B LDS store ---
    {
      v16h bv = {};
      if (bvalid) {
        int kidx = k0 + kkb;
        int kend = K - kidx;            // elements remaining
        int ci = kidx / khw;
        int r  = kidx - ci * khw;
        int ky = r / kw;
        int kx = r - ky * kw;
        const float* inb = in + binbase;
#pragma unroll
        for (int q = 0; q < 16; ++q) {
          if (q < kend) {
            int iy = boy * stride - pad + ky * dil;
            int ix = box * stride - pad + kx * dil;
            if (iy >= 0 && iy < H && ix >= 0 && ix < W)
              bv[q] = (_Float16)inb[((long long)ci * H + iy) * W + ix];
          }
          ++kx;
          if (kx == kw) { kx = 0; ++ky; if (ky == kh) { ky = 0; ++ci; } }
        }
      }
      *(v16h*)(Bs + (bcol << 5) + kkb) = bv;
    }
    __syncthreads();

    // --- fragments: 4 aligned 32B LDS vector loads per lane ---
    const int rl = lane & 15;
    const int hi = (lane >= 16) ? 1 : 0;
    const v16h* Af = (const v16h*)As;
    const v16h* Bf = (const v16h*)Bs;
    v16h a0 = Af[((waveM + rl) << 1) | hi];
    v16h a1 = Af[((waveM + rl + 16) << 1) | hi];
    v16h b0 = Bf[((waveN + rl) << 1) | hi];
    v16h b1 = Bf[((waveN + rl + 16) << 1) | hi];

    accs[0][0] = __builtin_amdgcn_wmma_f32_16x16x32_f16(false, a0, false, b0,
                                                        (short)0, accs[0][0], false, false);
    accs[0][1] = __builtin_amdgcn_wmma_f32_16x16x32_f16(false, a0, false, b1,
                                                        (short)0, accs[0][1], false, false);
    accs[1][0] = __builtin_amdgcn_wmma_f32_16x16x32_f16(false, a1, false, b0,
                                                        (short)0, accs[1][0], false, false);
    accs[1][1] = __builtin_amdgcn_wmma_f32_16x16x32_f16(false, a1, false, b1,
                                                        (short)0, accs[1][1], false, false);
    __syncthreads();
  }

  // --- epilogue: C/D layout: VGPR r -> M = r (+8 for hi lanes), N = lane%16
  const int nn   = lane & 15;
  const int mofs = (lane >= 16) ? 8 : 0;
#pragma unroll
  for (int i2 = 0; i2 < 2; ++i2) {
#pragma unroll
    for (int j2 = 0; j2 < 2; ++j2) {
#pragma unroll
      for (int r = 0; r < 8; ++r) {
        int co  = coutBase + waveM + i2 * 16 + r + mofs;
        int pix = pixBase + waveN + j2 * 16 + nn;
        if (co < Cout && pix < npix) {
          float v = accs[i2][j2][r] + bias[co];
          if (relu) v = fmaxf(v, 0.f);
          int b   = pix / HoWo;
          int rem = pix - b * HoWo;
          if (nhwcTotal > 0) {
            out[(long long)b * nhwcTotal + nhwcOff + rem * Cout + co] = v;
          } else {
            int oy = rem / Wo;
            int ox = rem - oy * Wo;
            out[(((long long)b * Cout + co) * Ho + oy) * Wo + ox] = v;
          }
        }
      }
    }
  }
}

// ---------------------------------------------------------------------------
// Max pool (asymmetric high-side padding handled by OOB skip)
// ---------------------------------------------------------------------------
__global__ void maxpool_kernel(const float* __restrict__ in, float* __restrict__ out,
                               int B, int C, int H, int W,
                               int k, int s, int pad, int Ho, int Wo) {
  int i = blockIdx.x * 256 + threadIdx.x;
  int total = B * C * Ho * Wo;
  if (i >= total) return;
  int ox = i % Wo; int t = i / Wo;
  int oy = t % Ho; t /= Ho;
  int c  = t % C;  int b = t / C;
  float m = -1e30f;
  for (int ky = 0; ky < k; ++ky) {
    int iy = oy * s - pad + ky;
    if (iy < 0 || iy >= H) continue;
    for (int kx = 0; kx < k; ++kx) {
      int ix = ox * s - pad + kx;
      if (ix < 0 || ix >= W) continue;
      m = fmaxf(m, in[(((long long)b * C + c) * H + iy) * W + ix]);
    }
  }
  out[i] = m;
}

// ---------------------------------------------------------------------------
// L2 norm across channels, scaled by per-channel weight
// ---------------------------------------------------------------------------
__global__ void l2norm_kernel(const float* __restrict__ in, const float* __restrict__ w,
                              float* __restrict__ out, int B, int C, int HW) {
  int i = blockIdx.x * 256 + threadIdx.x;
  if (i >= B * HW) return;
  int b = i / HW, p = i - b * HW;
  const float* base = in + (long long)b * C * HW + p;
  float ss = 0.f;
  for (int c = 0; c < C; ++c) { float v = base[(long long)c * HW]; ss += v * v; }
  float inv = 1.f / (sqrtf(ss) + 1e-10f);
  float* ob = out + (long long)b * C * HW + p;
  for (int c = 0; c < C; ++c) ob[(long long)c * HW] = w[c] * base[(long long)c * HW] * inv;
}

// ---------------------------------------------------------------------------
// softmax over 21 classes, in place
// ---------------------------------------------------------------------------
__global__ void softmax21_kernel(float* __restrict__ conf, int n) {
  int i = blockIdx.x * 256 + threadIdx.x;
  if (i >= n) return;
  float* p = conf + (long long)i * NCLS;
  float mx = p[0];
  for (int c = 1; c < NCLS; ++c) mx = fmaxf(mx, p[c]);
  float e[NCLS], s = 0.f;
  for (int c = 0; c < NCLS; ++c) { e[c] = __expf(p[c] - mx); s += e[c]; }
  float inv = 1.f / s;
  for (int c = 0; c < NCLS; ++c) p[c] = e[c] * inv;
}

// ---------------------------------------------------------------------------
// decode loc vs priors -> corner boxes
// ---------------------------------------------------------------------------
__global__ void decode_kernel(const float* __restrict__ loc,
                              const float* __restrict__ priors,
                              float* __restrict__ boxes, int n) {
  int i = blockIdx.x * 256 + threadIdx.x;
  if (i >= n) return;
  int pr = i % NUMPRIORS;
  const float* l = loc + (long long)i * 4;
  const float* p = priors + (long long)pr * 4;
  float cx = p[0] + l[0] * 0.1f * p[2];
  float cy = p[1] + l[1] * 0.1f * p[3];
  float w  = p[2] * __expf(l[2] * 0.2f);
  float h  = p[3] * __expf(l[3] * 0.2f);
  float* o = boxes + (long long)i * 4;
  o[0] = cx - w * 0.5f; o[1] = cy - h * 0.5f;
  o[2] = cx + w * 0.5f; o[3] = cy + h * 0.5f;
}

__global__ void zero_kernel(float* p, int n) {
  int i = blockIdx.x * 256 + threadIdx.x;
  if (i < n) p[i] = 0.f;
}

// ---------------------------------------------------------------------------
// per (batch, class) top-200 + NMS + stable compaction
// ---------------------------------------------------------------------------
__global__ __launch_bounds__(256)
void detect_kernel(const float* __restrict__ conf, const float* __restrict__ boxes,
                   float* __restrict__ out) {
  __shared__ float s[NUMPRIORS];
  __shared__ float rmax[256];
  __shared__ int   ridx[256];
  __shared__ float ts[TOPK];
  __shared__ int   ti[TOPK];
  __shared__ float bx[TOPK * 4];
  __shared__ float area[TOPK];
  __shared__ int   keep[TOPK];
  __shared__ int   pos[TOPK];

  const int tid = threadIdx.x;
  const int cls = blockIdx.x + 1;  // classes 1..20 (0 = background, all zeros)
  const int b   = blockIdx.y;

  for (int i = tid; i < NUMPRIORS; i += 256) {
    float v = conf[((long long)b * NUMPRIORS + i) * NCLS + cls];
    s[i] = (v > CONF_TH) ? v : -1.0f;
  }
  __syncthreads();

  // iterative top-K (block argmax)
  for (int t = 0; t < TOPK; ++t) {
    float bv = -1e30f; int bi = 0;
    for (int i = tid; i < NUMPRIORS; i += 256) {
      float v = s[i];
      if (v > bv) { bv = v; bi = i; }
    }
    rmax[tid] = bv; ridx[tid] = bi;
    __syncthreads();
    for (int off = 128; off > 0; off >>= 1) {
      if (tid < off) {
        float v2 = rmax[tid + off];
        if (v2 > rmax[tid] || (v2 == rmax[tid] && ridx[tid + off] < ridx[tid])) {
          rmax[tid] = v2; ridx[tid] = ridx[tid + off];
        }
      }
      __syncthreads();
    }
    if (tid == 0) { ts[t] = rmax[0]; ti[t] = ridx[0]; s[ridx[0]] = -2.0f; }
    __syncthreads();
  }

  // gather candidate boxes, areas, validity
  if (tid < TOPK) {
    const float* bb = boxes + ((long long)b * NUMPRIORS + ti[tid]) * 4;
    float x1 = bb[0], y1 = bb[1], x2 = bb[2], y2 = bb[3];
    bx[tid * 4 + 0] = x1; bx[tid * 4 + 1] = y1;
    bx[tid * 4 + 2] = x2; bx[tid * 4 + 3] = y2;
    area[tid] = (x2 - x1) * (y2 - y1);
    keep[tid] = (ts[tid] > CONF_TH) ? 1 : 0;
  }
  __syncthreads();

  // NMS (sequential over i, parallel over j > i)
  for (int i = 0; i < TOPK; ++i) {
    if (keep[i] && tid < TOPK && tid > i) {
      float xx1 = fmaxf(bx[i * 4 + 0], bx[tid * 4 + 0]);
      float yy1 = fmaxf(bx[i * 4 + 1], bx[tid * 4 + 1]);
      float xx2 = fminf(bx[i * 4 + 2], bx[tid * 4 + 2]);
      float yy2 = fminf(bx[i * 4 + 3], bx[tid * 4 + 3]);
      float inter = fmaxf(xx2 - xx1, 0.f) * fmaxf(yy2 - yy1, 0.f);
      float iou = inter / fmaxf(area[i] + area[tid] - inter, 1e-10f);
      if (iou > NMS_TH) keep[tid] = 0;
    }
    __syncthreads();
  }

  // stable compaction positions (kept boxes first, original order)
  if (tid == 0) {
    int c = 0;
    for (int j = 0; j < TOPK; ++j) { pos[j] = c; if (keep[j]) c++; }
  }
  __syncthreads();

  float* obase = out + ((long long)b * NCLS + cls) * TOPK * 5;
  for (int i = tid; i < TOPK * 5; i += 256) obase[i] = 0.f;
  __syncthreads();
  if (tid < TOPK && keep[tid]) {
    float* o = obase + pos[tid] * 5;
    o[0] = ts[tid];
    o[1] = bx[tid * 4 + 0]; o[2] = bx[tid * 4 + 1];
    o[3] = bx[tid * 4 + 2]; o[4] = bx[tid * 4 + 3];
  }
}

// ---------------------------------------------------------------------------
// host orchestration
// ---------------------------------------------------------------------------
extern "C" void kernel_launch(void* const* d_in, const int* in_sizes, int n_in,
                              void* d_out, int out_size, void* d_ws, size_t ws_size,
                              hipStream_t stream) {
  (void)in_sizes; (void)out_size; (void)ws_size;
  const float* x      = (const float*)d_in[0];
  const float* l2w    = (const float*)d_in[n_in - 2];   // (1,512,1,1) -> 512
  const float* priors = (const float*)d_in[n_in - 1];   // 8732 x 4

  struct Spec { int cin, cout, k, stride, pad, dil, relu; };
  static const Spec S[35] = {
    {3,64,3,1,1,1,1},    {64,64,3,1,1,1,1},                         // v1
    {64,128,3,1,1,1,1},  {128,128,3,1,1,1,1},                       // v2
    {128,256,3,1,1,1,1}, {256,256,3,1,1,1,1}, {256,256,3,1,1,1,1},  // v3
    {256,512,3,1,1,1,1}, {512,512,3,1,1,1,1}, {512,512,3,1,1,1,1},  // v4
    {512,512,3,1,1,1,1}, {512,512,3,1,1,1,1}, {512,512,3,1,1,1,1},  // v5
    {512,1024,3,1,6,6,1},{1024,1024,1,1,0,1,1},                     // conv6/7
    {1024,256,1,1,0,1,1},{256,512,3,2,1,1,1},                       // e1/e2
    {512,128,1,1,0,1,1}, {128,256,3,2,1,1,1},                       // e3/e4
    {256,128,1,1,0,1,1}, {128,256,3,1,0,1,1},                       // e5/e6
    {256,128,1,1,0,1,1}, {128,256,3,1,0,1,1},                       // e7/e8
    {512,16,3,1,1,1,0},  {1024,24,3,1,1,1,0}, {512,24,3,1,1,1,0},   // loc1-3
    {256,24,3,1,1,1,0},  {256,16,3,1,1,1,0},  {256,16,3,1,1,1,0},   // loc4-6
    {512,84,3,1,1,1,0},  {1024,126,3,1,1,1,0},{512,126,3,1,1,1,0},  // conf1-3
    {256,126,3,1,1,1,0}, {256,84,3,1,1,1,0},  {256,84,3,1,1,1,0},   // conf4-6
  };

  // workspace partition
  char* wsb = (char*)d_ws;
  size_t off = 0;
  auto alloc = [&](size_t bytes) -> char* {
    char* p = wsb + off;
    off += (bytes + 255) & ~(size_t)255;
    return p;
  };
  float*    bufA   = (float*)alloc((size_t)NB * 64 * 300 * 300 * 4);
  float*    bufB   = (float*)alloc((size_t)NB * 64 * 300 * 300 * 4);
  _Float16* w16    = (_Float16*)alloc((size_t)27000000 * 2);
  float*    s1     = (float*)alloc((size_t)NB * 512 * 38 * 38 * 4);
  float*    s2     = (float*)alloc((size_t)NB * 1024 * 19 * 19 * 4);
  float*    s3     = (float*)alloc((size_t)NB * 512 * 10 * 10 * 4);
  float*    s4     = (float*)alloc((size_t)NB * 256 * 5 * 5 * 4);
  float*    s5     = (float*)alloc((size_t)NB * 256 * 3 * 3 * 4);
  float*    s6     = (float*)alloc((size_t)NB * 256 * 1 * 1 * 4);
  float*    locbuf = (float*)alloc((size_t)NB * 34928 * 4);
  float*    confb  = (float*)alloc((size_t)NB * 183372 * 4);
  float*    boxes  = (float*)alloc((size_t)NB * NUMPRIORS * 4 * 4);

  // padded weight strides, offsets, fp16 packing
  int Kpad[35];
  size_t wofs[35]; size_t acc = 0;
  for (int i = 0; i < 35; ++i) {
    int K = S[i].cin * S[i].k * S[i].k;
    Kpad[i] = (K + 31) & ~31;
    wofs[i] = acc;
    acc += (size_t)S[i].cout * Kpad[i];
  }
  for (int i = 0; i < 35; ++i) {
    int K = S[i].cin * S[i].k * S[i].k;
    int n = S[i].cout * Kpad[i];
    pack_w16_kernel<<<(n + 255) / 256, 256, 0, stream>>>(
        (const float*)d_in[1 + 2 * i], w16 + wofs[i], S[i].cout, K, Kpad[i]);
  }

  auto conv = [&](int li, const float* in, float* out, int H, int W,
                  int nhwcTotal = 0, int nhwcOff = 0) {
    const Spec& sp = S[li];
    int Ho = (H + 2 * sp.pad - sp.dil * (sp.k - 1) - 1) / sp.stride + 1;
    int Wo = (W + 2 * sp.pad - sp.dil * (sp.k - 1) - 1) / sp.stride + 1;
    int npix = NB * Ho * Wo;
    dim3 grid((sp.cout + 63) / 64, (npix + 63) / 64);
    conv_wmma_kernel<<<grid, 128, 0, stream>>>(
        in, w16 + wofs[li], (const float*)d_in[2 + 2 * li], out,
        NB, sp.cin, H, W, sp.cout, Ho, Wo, sp.k, sp.k, sp.stride, sp.pad, sp.dil,
        sp.relu, Kpad[li], nhwcTotal, nhwcOff);
  };
  auto pool = [&](const float* in, float* out, int C, int H, int W,
                  int k, int s, int pad, int extra) {
    int Ho = (H + 2 * pad + extra - k) / s + 1;
    int Wo = (W + 2 * pad + extra - k) / s + 1;
    int total = NB * C * Ho * Wo;
    maxpool_kernel<<<(total + 255) / 256, 256, 0, stream>>>(in, out, NB, C, H, W,
                                                            k, s, pad, Ho, Wo);
  };

  // ---- backbone ----
  conv(0, x, bufA, 300, 300);
  conv(1, bufA, bufB, 300, 300);
  pool(bufB, bufA, 64, 300, 300, 2, 2, 0, 0);        // -> 150
  conv(2, bufA, bufB, 150, 150);
  conv(3, bufB, bufA, 150, 150);
  pool(bufA, bufB, 128, 150, 150, 2, 2, 0, 0);       // -> 75
  conv(4, bufB, bufA, 75, 75);
  conv(5, bufA, bufB, 75, 75);
  conv(6, bufB, bufA, 75, 75);
  pool(bufA, bufB, 256, 75, 75, 2, 2, 0, 1);         // -> 38 (extra pad)
  conv(7, bufB, bufA, 38, 38);
  conv(8, bufA, bufB, 38, 38);
  conv(9, bufB, bufA, 38, 38);                       // v4_3 in bufA
  {
    int n = NB * 38 * 38;
    l2norm_kernel<<<(n + 255) / 256, 256, 0, stream>>>(bufA, l2w, s1, NB, 512,
                                                       38 * 38);
  }
  pool(bufA, bufB, 512, 38, 38, 2, 2, 0, 0);         // -> 19
  conv(10, bufB, bufA, 19, 19);
  conv(11, bufA, bufB, 19, 19);
  conv(12, bufB, bufA, 19, 19);
  pool(bufA, bufB, 512, 19, 19, 3, 1, 1, 0);         // -> 19
  conv(13, bufB, bufA, 19, 19);                      // conv6
  conv(14, bufA, s2, 19, 19);                        // conv7 -> s2
  conv(15, s2, bufA, 19, 19);                        // e1
  conv(16, bufA, s3, 19, 19);                        // e2 -> s3 (10x10)
  conv(17, s3, bufA, 10, 10);                        // e3
  conv(18, bufA, s4, 10, 10);                        // e4 -> s4 (5x5)
  conv(19, s4, bufA, 5, 5);                          // e5
  conv(20, bufA, s5, 5, 5);                          // e6 -> s5 (3x3)
  conv(21, s5, bufA, 3, 3);                          // e7
  conv(22, bufA, s6, 3, 3);                          // e8 -> s6 (1x1)

  // ---- heads: conv writes NHWC-concat directly ----
  struct Head { const float* src; int Hh, locIdx, confIdx, locOff, confOff; };
  const Head heads[6] = {
    {s1, 38, 23, 29,     0,      0},
    {s2, 19, 24, 30, 23104, 121296},
    {s3, 10, 25, 31, 31768, 166782},
    {s4,  5, 26, 32, 34168, 179382},
    {s5,  3, 27, 33, 34768, 182532},
    {s6,  1, 28, 34, 34912, 183288},
  };
  const int LOC_TOT = 34928, CONF_TOT = 183372;
  for (int h = 0; h < 6; ++h) {
    const Head& hd = heads[h];
    conv(hd.locIdx,  hd.src, locbuf, hd.Hh, hd.Hh, LOC_TOT,  hd.locOff);
    conv(hd.confIdx, hd.src, confb,  hd.Hh, hd.Hh, CONF_TOT, hd.confOff);
  }

  // ---- tail: softmax, decode, detect ----
  {
    int n = NB * NUMPRIORS;
    softmax21_kernel<<<(n + 255) / 256, 256, 0, stream>>>(confb, n);
    decode_kernel<<<(n + 255) / 256, 256, 0, stream>>>(locbuf, priors, boxes, n);
  }
  {
    int n = NB * NCLS * TOPK * 5;
    zero_kernel<<<(n + 255) / 256, 256, 0, stream>>>((float*)d_out, n);
  }
  dim3 dg(NCLS - 1, NB);
  detect_kernel<<<dg, 256, 0, stream>>>(confb, boxes, (float*)d_out);
}
